// AttentionSup_37142877176222
// MI455X (gfx1250) — matrix operand order repaired and checked by reading, not compiled, not run on hardware
//
#include <hip/hip_runtime.h>
#include <hip/hip_bf16.h>

// ---------------------------------------------------------------------------
// Types for WMMA fragments (wave32, gfx1250)
// ---------------------------------------------------------------------------
typedef _Float16 v16h __attribute__((ext_vector_type(16)));
typedef _Float16 v8h  __attribute__((ext_vector_type(8)));
typedef float    v8f  __attribute__((ext_vector_type(8)));

#define DIMC   512
#define HEADS  8
#define DHEAD  64
#define INNER  512
#define NSEQ   2048
#define BATCH  4
#define QKV_ELEMS (BATCH * HEADS * NSEQ * DHEAD)   // 4,194,304 halves per tensor

// ---------------------------------------------------------------------------
// Tensor Data Mover support (gfx1250). Guarded: falls back to manual copies.
// ---------------------------------------------------------------------------
#if defined(__has_builtin)
#if __has_builtin(__builtin_amdgcn_tensor_load_to_lds) && \
    __has_builtin(__builtin_amdgcn_s_wait_tensorcnt)
#define USE_TDM 1
#endif
#endif
#ifndef USE_TDM
#define USE_TDM 0
#endif

#if USE_TDM
typedef unsigned int v4u __attribute__((ext_vector_type(4)));
typedef int          v4i __attribute__((ext_vector_type(4)));
typedef int          v8i __attribute__((ext_vector_type(8)));

__device__ __forceinline__ unsigned lds_off32(const void* p) {
  // generic pointer to LDS: addr[31:0] is the LDS byte offset (aperture rule)
  return (unsigned)(unsigned long long)(size_t)p;
}

// 2D f16 tile load via TDM: tile_d0 x tile_d1 elements (2B each), global row
// stride g_stride elements, LDS padding pad codes per D# group1 encoding.
__device__ __forceinline__ void tdm_load_2d_f16(
    const _Float16* gsrc, unsigned lds_byte_off,
    unsigned tile_d0, unsigned tile_d1, unsigned g_stride,
    unsigned tdim0, unsigned tdim1,
    unsigned pad_interval_code, unsigned pad_amount_code) {
  unsigned long long ga = (unsigned long long)(size_t)gsrc;
  v4u g0;
  g0[0] = 1u;                                        // count=1 (valid user D#)
  g0[1] = lds_byte_off;                              // lds_addr
  g0[2] = (unsigned)(ga & 0xffffffffu);              // global_addr[31:0]
  g0[3] = (unsigned)((ga >> 32) & 0x01ffffffu)       // global_addr[56:32]
        | (2u << 30);                                // type=2 ("image")
  v8i g1;
  g1[0] = (int)((1u << 16)                           // data_size=1 -> 2 bytes
        | (1u << 20)                                 // pad_enable
        | (pad_interval_code << 22)
        | (pad_amount_code << 25));
  g1[1] = (int)((tdim0 & 0xffffu) << 16);            // tensor_dim0[15:0]
  g1[2] = (int)((tdim0 >> 16) | ((tdim1 & 0xffffu) << 16));
  g1[3] = (int)(((tdim1 >> 16) & 0xffffu) | (tile_d0 << 16));  // tile_dim0
  g1[4] = (int)(tile_d1 & 0xffffu);                  // tile_dim1 (tile_dim2=0)
  g1[5] = (int)g_stride;                             // tensor_dim0_stride lo
  g1[6] = 0;
  g1[7] = 0;
  v4i z = {0, 0, 0, 0};
#if __clang_major__ >= 23
  v8i z8 = {0, 0, 0, 0, 0, 0, 0, 0};
  __builtin_amdgcn_tensor_load_to_lds(g0, g1, z, z, z8, 0);
#else
  __builtin_amdgcn_tensor_load_to_lds(g0, g1, z, z, 0);
#endif
}
#endif  // USE_TDM

// ---------------------------------------------------------------------------
// LDS fragment loaders per CDNA5 ISA 16-bit A(16x32)/B(32x16) layouts.
// ---------------------------------------------------------------------------
__device__ __forceinline__ v16h ldsA16(const _Float16* p, int row0, int stride,
                                       int k0, int lane) {
  int r  = lane & 15;
  int hi = (lane >> 4) << 3;  // 0 or 8
  const _Float16* base = p + (row0 + r) * stride + k0 + hi;
  v8h lo = *(const v8h*)(base);
  v8h hh = *(const v8h*)(base + 16);
  return __builtin_shufflevector(lo, hh, 0,1,2,3,4,5,6,7,8,9,10,11,12,13,14,15);
}

__device__ __forceinline__ v16h ldsB16(const _Float16* p, int col0, int stride,
                                       int k0, int lane) {
  int c  = lane & 15;
  int hi = (lane >> 4) << 3;  // 0 or 8
  const _Float16* base = p + (col0 + c) * stride + k0 + hi;
  v8h lo = *(const v8h*)(base);
  v8h hh = *(const v8h*)(base + 16);
  return __builtin_shufflevector(lo, hh, 0,1,2,3,4,5,6,7,8,9,10,11,12,13,14,15);
}

__device__ __forceinline__ v8f wmma_f16(v16h a, v16h b, v8f c) {
  return __builtin_amdgcn_wmma_f32_16x16x32_f16(false, a, false, b,
                                                (short)0, c, false, false);
}

__device__ __forceinline__ v8h pack8(float4 f0, float4 f1) {
  v8h h;
  h[0] = (_Float16)f0.x; h[1] = (_Float16)f0.y;
  h[2] = (_Float16)f0.z; h[3] = (_Float16)f0.w;
  h[4] = (_Float16)f1.x; h[5] = (_Float16)f1.y;
  h[6] = (_Float16)f1.z; h[7] = (_Float16)f1.w;
  return h;
}

// ---------------------------------------------------------------------------
// Kernel 1: domain gate MLP  gate[b,h,k] = softmax_h( relu(dl@W1+b1)@W2+b2 )
// ---------------------------------------------------------------------------
__global__ __launch_bounds__(256) void gate_kernel(
    const float* __restrict__ dl, const float* __restrict__ Wd1,
    const float* __restrict__ bd1, const float* __restrict__ Wd2,
    const float* __restrict__ bd2, float* __restrict__ gate) {
  int t = threadIdx.x;
  int b = t >> 6;
  int k = t & 63;
  float hid[32];
#pragma unroll
  for (int h = 0; h < 32; ++h) {
    float a = bd1[h];
#pragma unroll
    for (int j = 0; j < 4; ++j) a += dl[b * 4 + j] * Wd1[j * 32 + h];
    hid[h] = a > 0.f ? a : 0.f;
  }
  float vals[8];
  float mx = -1e30f;
#pragma unroll
  for (int hd = 0; hd < 8; ++hd) {
    int col = hd * 64 + k;
    float a = bd2[col];
    for (int h = 0; h < 32; ++h) a += hid[h] * Wd2[h * 512 + col];
    vals[hd] = a;
    mx = fmaxf(mx, a);
  }
  float s = 0.f;
#pragma unroll
  for (int hd = 0; hd < 8; ++hd) { vals[hd] = __expf(vals[hd] - mx); s += vals[hd]; }
  float inv = 1.f / s;
#pragma unroll
  for (int hd = 0; hd < 8; ++hd) gate[b * 512 + hd * 64 + k] = vals[hd] * inv;
}

// ---------------------------------------------------------------------------
// Kernel 2: QKV projection GEMM  [8192,512] x [512,1536] -> f16 q/k/v [b,h,n,d]
// ---------------------------------------------------------------------------
#define LDA 40
#define LDB 40

__global__ __launch_bounds__(256) void qkv_kernel(
    const float* __restrict__ x, const float* __restrict__ W,
    _Float16* __restrict__ qkv) {
  __shared__ alignas(16) _Float16 aS[128 * LDA];
  __shared__ alignas(16) _Float16 bS[64 * LDB];
  int tid = threadIdx.x;
  int lane = tid & 31, wave = tid >> 5;
  int wm = wave & 3, wn = wave >> 2;
  int r0 = blockIdx.y * 128;
  int c0 = blockIdx.x * 64;

  v8f acc[2][2] = {};

  for (int kt = 0; kt < DIMC; kt += 32) {
    {  // A tile: 128x32 f32 -> f16, packed 16B LDS stores
      int row = tid >> 1;
      int kp = (tid & 1) * 16;
      const float* src = x + (size_t)(r0 + row) * DIMC + kt + kp;
      float4 f0 = *(const float4*)(src);
      float4 f1 = *(const float4*)(src + 4);
      float4 f2 = *(const float4*)(src + 8);
      float4 f3 = *(const float4*)(src + 12);
      *(v8h*)(aS + row * LDA + kp) = pack8(f0, f1);
      *(v8h*)(aS + row * LDA + kp + 8) = pack8(f2, f3);
    }
    {  // B tile: 32x64 f32 -> f16, stored col-major bt[n][k]
      int k = tid >> 3;
      int cg = (tid & 7) * 8;
      const float* src = W + (size_t)(kt + k) * 1536 + c0 + cg;
      float4 f0 = *(const float4*)src;
      float4 f1 = *(const float4*)(src + 4);
      bS[(cg + 0) * LDB + k] = (_Float16)f0.x;
      bS[(cg + 1) * LDB + k] = (_Float16)f0.y;
      bS[(cg + 2) * LDB + k] = (_Float16)f0.z;
      bS[(cg + 3) * LDB + k] = (_Float16)f0.w;
      bS[(cg + 4) * LDB + k] = (_Float16)f1.x;
      bS[(cg + 5) * LDB + k] = (_Float16)f1.y;
      bS[(cg + 6) * LDB + k] = (_Float16)f1.z;
      bS[(cg + 7) * LDB + k] = (_Float16)f1.w;
    }
    __syncthreads();
    v16h a0 = ldsA16(aS, 32 * wm, LDA, 0, lane);
    v16h a1 = ldsA16(aS, 32 * wm + 16, LDA, 0, lane);
    v16h b0 = ldsB16(bS, 32 * wn, LDB, 0, lane);
    v16h b1 = ldsB16(bS, 32 * wn + 16, LDB, 0, lane);
    acc[0][0] = wmma_f16(a0, b0, acc[0][0]);
    acc[0][1] = wmma_f16(a0, b1, acc[0][1]);
    acc[1][0] = wmma_f16(a1, b0, acc[1][0]);
    acc[1][1] = wmma_f16(a1, b1, acc[1][1]);
    __syncthreads();
  }

  int part = c0 >> 9;
  int h = (c0 & 511) >> 6;
  _Float16* dst = qkv + (size_t)part * QKV_ELEMS;
#pragma unroll
  for (int mi = 0; mi < 2; ++mi)
#pragma unroll
    for (int ni = 0; ni < 2; ++ni)
#pragma unroll
      for (int j = 0; j < 8; ++j) {
        int row = r0 + 32 * wm + 16 * mi + ((lane >> 4) << 3) + j;
        int dh = 32 * wn + 16 * ni + (lane & 15);
        int b = row >> 11, n = row & 2047;
        dst[(((size_t)b * HEADS + h) * NSEQ + n) * DHEAD + dh] =
            (_Float16)acc[mi][ni][j];
      }
}

// ---------------------------------------------------------------------------
// Kernel 3: flash attention per (b,h): Br=Bc=64, online softmax, gate epilogue.
// Q/K tiles arrive via TDM (pad 64->72 half rows); V transposed manually.
// ---------------------------------------------------------------------------
__global__ __launch_bounds__(256) void flash_kernel(
    const _Float16* __restrict__ qg, const _Float16* __restrict__ kg,
    const _Float16* __restrict__ vg, const float* __restrict__ gate,
    _Float16* __restrict__ attn) {
  __shared__ alignas(16) _Float16 qS[64 * 72];
  __shared__ alignas(16) _Float16 kS[64 * 72];   // [m][d]
  __shared__ alignas(16) _Float16 vS[64 * 72];   // transposed [d][m]
  __shared__ alignas(16) _Float16 pS[64 * 72];   // [row][m]
  __shared__ alignas(16) float sS[64 * 65];
  __shared__ float mSt[64], lSt[64], aSt[64];

  int tid = threadIdx.x, lane = tid & 31, wave = tid >> 5;
  int wm = wave & 3, wn = wave >> 2;
  int qb = blockIdx.x, h = blockIdx.y, b = blockIdx.z;
  int n0 = qb * 64;
  size_t headBase = ((size_t)b * HEADS + h) * NSEQ * DHEAD;

#if USE_TDM
  if (wave == 0)
    tdm_load_2d_f16(qg + headBase + (size_t)n0 * DHEAD, lds_off32(qS),
                    64, 64, 64, 64, 2048, /*padI=*/4, /*padA=*/3);
#else
  {  // manual Q tile
    int row = tid >> 2, dc = (tid & 3) * 16;
    const v8h* s = (const v8h*)(qg + headBase + (size_t)(n0 + row) * DHEAD + dc);
    *(v8h*)(qS + row * 72 + dc) = s[0];
    *(v8h*)(qS + row * 72 + dc + 8) = s[1];
  }
#endif
  if (tid < 64) { mSt[tid] = -1e30f; lSt[tid] = 0.f; }
  v8f accO[2] = {};
#if USE_TDM
  __builtin_amdgcn_s_wait_tensorcnt(0);
#endif
  __syncthreads();

  for (int mb = 0; mb < NSEQ; mb += 64) {
#if USE_TDM
    if (wave == 0)
      tdm_load_2d_f16(kg + headBase + (size_t)mb * DHEAD, lds_off32(kS),
                      64, 64, 64, 64, 2048, 4, 3);
#else
    {  // manual K tile [m][d]
      int row = tid >> 2, dc = (tid & 3) * 16;
      const v8h* s = (const v8h*)(kg + headBase + (size_t)(mb + row) * DHEAD + dc);
      *(v8h*)(kS + row * 72 + dc) = s[0];
      *(v8h*)(kS + row * 72 + dc + 8) = s[1];
    }
#endif
    {  // V tile transposed -> vS[d][m]
      int m = tid >> 2, dc = (tid & 3) * 16;
      const _Float16* s = vg + headBase + (size_t)(mb + m) * DHEAD + dc;
#pragma unroll
      for (int i = 0; i < 16; ++i) vS[(dc + i) * 72 + m] = s[i];
    }
#if USE_TDM
    __builtin_amdgcn_s_wait_tensorcnt(0);
#endif
    __syncthreads();

    // S = Q @ K^T  (K-dim = DHEAD = 2 WMMA K-steps)
    v8f sa[2] = {};
#pragma unroll
    for (int ks = 0; ks < 64; ks += 32) {
      v16h a = ldsA16(qS, 16 * wm, 72, ks, lane);
#pragma unroll
      for (int ni = 0; ni < 2; ++ni) {
        v16h bf = ldsB16(kS, 16 * (2 * wn + ni), 72, ks, lane);
        sa[ni] = wmma_f16(a, bf, sa[ni]);
      }
    }
#pragma unroll
    for (int ni = 0; ni < 2; ++ni)
#pragma unroll
      for (int j = 0; j < 8; ++j) {
        int row = 16 * wm + ((lane >> 4) << 3) + j;
        int col = 16 * (2 * wn + ni) + (lane & 15);
        sS[row * 65 + col] = sa[ni][j] * 0.125f;  // DIM_HEAD^-0.5
      }
    __syncthreads();

    // online softmax: wave w owns rows 8w..8w+7
#pragma unroll
    for (int rr = 0; rr < 8; ++rr) {
      int row = 8 * wave + rr;
      float v0 = sS[row * 65 + lane];
      float v1 = sS[row * 65 + 32 + lane];
      float mx = fmaxf(v0, v1);
#pragma unroll
      for (int off = 16; off > 0; off >>= 1) mx = fmaxf(mx, __shfl_xor(mx, off, 32));
      float mOld = mSt[row];
      float mNew = fmaxf(mOld, mx);
      float p0 = __expf(v0 - mNew), p1 = __expf(v1 - mNew);
      pS[row * 72 + lane] = (_Float16)p0;
      pS[row * 72 + 32 + lane] = (_Float16)p1;
      float rs = p0 + p1;
#pragma unroll
      for (int off = 16; off > 0; off >>= 1) rs += __shfl_xor(rs, off, 32);
      if (lane == 0) {
        float al = __expf(mOld - mNew);
        aSt[row] = al;
        lSt[row] = lSt[row] * al + rs;
        mSt[row] = mNew;
      }
    }
    __syncthreads();

    // rescale O, accumulate P @ V
#pragma unroll
    for (int ni = 0; ni < 2; ++ni)
#pragma unroll
      for (int j = 0; j < 8; ++j) {
        int row = 16 * wm + ((lane >> 4) << 3) + j;
        accO[ni][j] *= aSt[row];
      }
#pragma unroll
    for (int ks = 0; ks < 64; ks += 32) {
      v16h a = ldsA16(pS, 16 * wm, 72, ks, lane);
#pragma unroll
      for (int ni = 0; ni < 2; ++ni) {
        v16h bf = ldsB16(vS, 16 * (2 * wn + ni), 72, ks, lane);
        accO[ni] = wmma_f16(a, bf, accO[ni]);
      }
    }
    __syncthreads();
  }

  // epilogue: normalize, apply head-gate, store f16 [b, n, h*64+dc]
#pragma unroll
  for (int ni = 0; ni < 2; ++ni) {
    int dc = 16 * (2 * wn + ni) + (lane & 15);
    float g = gate[(b * HEADS + h) * DHEAD + dc];
#pragma unroll
    for (int j = 0; j < 8; ++j) {
      int row = 16 * wm + ((lane >> 4) << 3) + j;
      float invl = 1.f / lSt[row];
      attn[((size_t)b * NSEQ + n0 + row) * INNER + h * DHEAD + dc] =
          (_Float16)(accO[ni][j] * invl * g);
    }
  }
}

// ---------------------------------------------------------------------------
// Kernel 4: output projection [8192,512](f16) x [512,512](f32->f16) + bias
// A tile (128x32 halves, 512-half rows) streamed by TDM with 32->40 padding.
// ---------------------------------------------------------------------------
__global__ __launch_bounds__(256) void out_kernel(
    const _Float16* __restrict__ attn, const float* __restrict__ W,
    const float* __restrict__ bias, float* __restrict__ out) {
  __shared__ alignas(16) _Float16 aS[128 * LDA];
  __shared__ alignas(16) _Float16 bS[64 * LDB];
  int tid = threadIdx.x;
  int lane = tid & 31, wave = tid >> 5;
  int wm = wave & 3, wn = wave >> 2;
  int r0 = blockIdx.y * 128;
  int c0 = blockIdx.x * 64;

  v8f acc[2][2] = {};

  for (int kt = 0; kt < INNER; kt += 32) {
#if USE_TDM
    if (wave == 0)
      tdm_load_2d_f16(attn + (size_t)r0 * INNER + kt, lds_off32(aS),
                      32, 128, 512, 32, 128, /*padI=*/3, /*padA=*/3);
#else
    {  // manual A tile (already f16)
      int row = tid >> 1;
      int kp = (tid & 1) * 16;
      const v8h* src = (const v8h*)(attn + (size_t)(r0 + row) * INNER + kt + kp);
      *(v8h*)(aS + row * LDA + kp) = src[0];
      *(v8h*)(aS + row * LDA + kp + 8) = src[1];
    }
#endif
    {  // B tile f32 -> f16, col-major
      int k = tid >> 3;
      int cg = (tid & 7) * 8;
      const float* src = W + (size_t)(kt + k) * DIMC + c0 + cg;
      float4 f0 = *(const float4*)src;
      float4 f1 = *(const float4*)(src + 4);
      bS[(cg + 0) * LDB + k] = (_Float16)f0.x;
      bS[(cg + 1) * LDB + k] = (_Float16)f0.y;
      bS[(cg + 2) * LDB + k] = (_Float16)f0.z;
      bS[(cg + 3) * LDB + k] = (_Float16)f0.w;
      bS[(cg + 4) * LDB + k] = (_Float16)f1.x;
      bS[(cg + 5) * LDB + k] = (_Float16)f1.y;
      bS[(cg + 6) * LDB + k] = (_Float16)f1.z;
      bS[(cg + 7) * LDB + k] = (_Float16)f1.w;
    }
#if USE_TDM
    __builtin_amdgcn_s_wait_tensorcnt(0);
#endif
    __syncthreads();
    v16h a0 = ldsA16(aS, 32 * wm, LDA, 0, lane);
    v16h a1 = ldsA16(aS, 32 * wm + 16, LDA, 0, lane);
    v16h b0 = ldsB16(bS, 32 * wn, LDB, 0, lane);
    v16h b1 = ldsB16(bS, 32 * wn + 16, LDB, 0, lane);
    acc[0][0] = wmma_f16(a0, b0, acc[0][0]);
    acc[0][1] = wmma_f16(a0, b1, acc[0][1]);
    acc[1][0] = wmma_f16(a1, b0, acc[1][0]);
    acc[1][1] = wmma_f16(a1, b1, acc[1][1]);
    __syncthreads();
  }

#pragma unroll
  for (int mi = 0; mi < 2; ++mi)
#pragma unroll
    for (int ni = 0; ni < 2; ++ni)
#pragma unroll
      for (int j = 0; j < 8; ++j) {
        int row = r0 + 32 * wm + 16 * mi + ((lane >> 4) << 3) + j;
        int c = c0 + 32 * wn + 16 * ni + (lane & 15);
        out[(size_t)row * DIMC + c] = acc[mi][ni][j] + bias[c];
      }
}

// ---------------------------------------------------------------------------
// Launch: gate -> qkv -> flash -> out
// ---------------------------------------------------------------------------
extern "C" void kernel_launch(void* const* d_in, const int* in_sizes, int n_in,
                              void* d_out, int out_size, void* d_ws, size_t ws_size,
                              hipStream_t stream) {
  (void)in_sizes; (void)n_in; (void)out_size; (void)ws_size;
  const float* x    = (const float*)d_in[0];
  const float* dl   = (const float*)d_in[1];
  const float* Wqkv = (const float*)d_in[2];
  const float* Wd1  = (const float*)d_in[3];
  const float* bd1  = (const float*)d_in[4];
  const float* Wd2  = (const float*)d_in[5];
  const float* bd2  = (const float*)d_in[6];
  const float* Wout = (const float*)d_in[7];
  const float* bout = (const float*)d_in[8];
  float* out = (float*)d_out;

  char* ws = (char*)d_ws;
  _Float16* qkv  = (_Float16*)ws;                       // 25,165,824 bytes
  float*    gate = (float*)(ws + 25165824);             // 8,192 bytes
  _Float16* attn = (_Float16*)(ws + 25174016);          // 8,388,608 bytes

  gate_kernel<<<1, 256, 0, stream>>>(dl, Wd1, bd1, Wd2, bd2, gate);
  qkv_kernel<<<dim3(24, 64), 256, 0, stream>>>(x, Wqkv, qkv);
  flash_kernel<<<dim3(32, 8, 4), 256, 0, stream>>>(
      qkv, qkv + QKV_ELEMS, qkv + 2 * (size_t)QKV_ELEMS, gate, attn);
  out_kernel<<<dim3(8, 64), 256, 0, stream>>>(attn, Wout, bout, out);
}
